// TJDGPT2OLD_42717744726823
// MI455X (gfx1250) — matrix operand order repaired and checked by reading, not compiled, not run on hardware
//
#include <hip/hip_runtime.h>
#include <hip/hip_bf16.h>

// ---------------- problem constants ----------------
static constexpr int kS   = 256;
static constexpr int kD   = 768;
static constexpr int kNH  = 12;
static constexpr int kDH  = 64;
static constexpr int kL   = 2;
static constexpr int kHOR = 8;
static constexpr int kTP  = kS - kHOR;        // 248
static constexpr int kR   = 2;
static constexpr int kV   = 50257;
static constexpr int kVR  = kV * kR;          // 100514
static constexpr int kRVR = kR * kV * kR;     // 201028 (core columns)
static constexpr int kTT  = 2 * kR + kRVR;    // 201032
static constexpr int kKS  = kD / 32;          // 24 k-steps
static constexpr float kEPS   = 1e-9f;
static constexpr float kLNEPS = 1e-5f;

typedef __attribute__((ext_vector_type(16))) _Float16 v16h;
typedef __attribute__((ext_vector_type(8)))  float    v8f;

// ---------------- wave reductions (wave32) ----------------
__device__ __forceinline__ float wred_sum(float v) {
#pragma unroll
  for (int o = 16; o > 0; o >>= 1) v += __shfl_xor(v, o, 32);
  return v;
}
__device__ __forceinline__ float wred_max(float v) {
#pragma unroll
  for (int o = 16; o > 0; o >>= 1) v = fmaxf(v, __shfl_xor(v, o, 32));
  return v;
}

// ---------------- WMMA fragment loaders (CDNA5 16x16x32 f16) ----------------
// A: 16x32 (MxK), fp32 row-major, converted to f16. Unguarded: caller
// guarantees rows [row0, row0+16) are readable (buffers padded to 16 rows).
// Lane l (l<16): row M=l, K-base 0; lane>=16: row M=l-16, K-base 16.
// VGPRs 0..7 hold K=kb..kb+15 -> one contiguous 16-float run
// => four global_load_b128.
__device__ __forceinline__ v16h load_a_frag(const float* __restrict__ A, int lda,
                                            int row0, int k0) {
  const int lane = threadIdx.x & 31;
  const int row  = row0 + (lane & 15);
  const int kb   = k0 + ((lane >> 4) << 4);
  const float* base = A + (size_t)row * lda + kb;
  const float4 x0 = *(const float4*)(base);
  const float4 x1 = *(const float4*)(base + 4);
  const float4 x2 = *(const float4*)(base + 8);
  const float4 x3 = *(const float4*)(base + 12);
  v16h f;
  f[0]  = (_Float16)x0.x; f[1]  = (_Float16)x0.y;
  f[2]  = (_Float16)x0.z; f[3]  = (_Float16)x0.w;
  f[4]  = (_Float16)x1.x; f[5]  = (_Float16)x1.y;
  f[6]  = (_Float16)x1.z; f[7]  = (_Float16)x1.w;
  f[8]  = (_Float16)x2.x; f[9]  = (_Float16)x2.y;
  f[10] = (_Float16)x2.z; f[11] = (_Float16)x2.w;
  f[12] = (_Float16)x3.x; f[13] = (_Float16)x3.y;
  f[14] = (_Float16)x3.z; f[15] = (_Float16)x3.w;
  return f;
}

// B: 32x16 (KxN), fp32 row-major (leading dim = N), converted to f16.
// Lane l: col N=l&15; lanes 0-15 hold K=k0..k0+15, lanes 16-31 K=k0+16..k0+31.
__device__ __forceinline__ v16h load_b_frag(const float* __restrict__ W, int ldw,
                                            int k0, int col0) {
  const int lane = threadIdx.x & 31;
  const int col  = col0 + (lane & 15);
  const int kb   = k0 + ((lane >> 4) << 4);
  v16h f;
#pragma unroll
  for (int i = 0; i < 16; ++i)
    f[i] = (_Float16)W[(size_t)(kb + i) * ldw + col];
  return f;
}

// B fragment from an LDS-staged f16 tile in COLUMN-major layout sW[c][k]:
// each lane's 16 halves are 32 contiguous bytes => two ds_load_b128.
__device__ __forceinline__ v16h load_b_frag_lds(const _Float16* sW, int k0) {
  const int lane = threadIdx.x & 31;
  const int col  = lane & 15;
  const int kb   = k0 + ((lane >> 4) << 4);
  return *(const v16h*)(sW + (size_t)col * kD + kb);
}

// Pre-packed f16 A fragments: lat16[rt][kstep][lane][16 halves].
__device__ __forceinline__ v16h load_a_packed(const _Float16* __restrict__ lat16,
                                              int rt, int kstep) {
  const int lane = threadIdx.x & 31;
  return *(const v16h*)(lat16 + (((size_t)rt * kKS + kstep) * 32 + lane) * 16);
}

// ---------------- embedding ----------------
__global__ void embed_kernel(const int* __restrict__ ids,
                             const float* __restrict__ wte,
                             const float* __restrict__ wpe,
                             float* __restrict__ h) {
  const int s   = blockIdx.x;
  const int tok = ids[s];
  for (int d = threadIdx.x; d < kD; d += blockDim.x)
    h[(size_t)s * kD + d] = wte[(size_t)tok * kD + d] + wpe[(size_t)s * kD + d];
}

// ---------------- layernorm (one block per row) ----------------
__global__ __launch_bounds__(256)
void ln_kernel(const float* __restrict__ x, const float* __restrict__ g,
               const float* __restrict__ b, float* __restrict__ y) {
  __shared__ float red[17];
  const int row = blockIdx.x, tid = threadIdx.x;
  const int lane = tid & 31, wid = tid >> 5;
  float s = 0.f, ss = 0.f;
  for (int d = tid; d < kD; d += 256) {
    float v = x[(size_t)row * kD + d];
    s += v; ss += v * v;
  }
  s = wred_sum(s); ss = wred_sum(ss);
  if (lane == 0) { red[wid] = s; red[8 + wid] = ss; }
  __syncthreads();
  if (tid == 0) {
    float S = 0.f, SS = 0.f;
    for (int i = 0; i < 8; ++i) { S += red[i]; SS += red[8 + i]; }
    float m = S / kD;
    red[16] = m;
    red[15] = SS / kD - m * m;
  }
  __syncthreads();
  const float m = red[16];
  const float inv = rsqrtf(red[15] + kLNEPS);
  for (int d = tid; d < kD; d += 256)
    y[(size_t)row * kD + d] = (x[(size_t)row * kD + d] - m) * inv * g[d] + b[d];
}

// ---------------- generic WMMA GEMM: C = epilogue(A @ W + bias) ----------------
// No row guards: all row counts are padded to multiples of 16 and all buffers
// are allocated with the padded row count. ACT: 0 none, 1 gelu, 2 relu.
template <int ACT>
__global__ void __launch_bounds__(128)
gemm_wmma(const float* __restrict__ A, const float* __restrict__ W,
          const float* __restrict__ bias, const float* __restrict__ residual,
          float* __restrict__ C, int N, int K) {
  const int wave = threadIdx.x >> 5;
  const int lane = threadIdx.x & 31;
  const int col0 = (blockIdx.x * 4 + wave) * 16;
  const int row0 = blockIdx.y * 16;

  v8f acc = {0.f, 0.f, 0.f, 0.f, 0.f, 0.f, 0.f, 0.f};
#pragma unroll 2
  for (int k = 0; k < K; k += 32) {
    if (k + 32 < K)  // hint next B tile into cache (global_prefetch_b8)
      __builtin_prefetch(&W[(size_t)(k + 32) * N + col0 + lane], 0, 0);
    v16h a = load_a_frag(A, K, row0, k);
    v16h b = load_b_frag(W, N, k, col0);
    acc = __builtin_amdgcn_wmma_f32_16x16x32_f16(false, a, false, b,
                                                 (short)0, acc, false, false);
  }
  const int col   = col0 + (lane & 15);
  const int rbase = row0 + ((lane >> 4) << 3);
  const float bv  = bias[col];
#pragma unroll
  for (int j = 0; j < 8; ++j) {
    float v = acc[j] + bv;
    if (ACT == 1) {
      float x3 = v * v * v;
      v = 0.5f * v * (1.f + tanhf(0.7978845608f * (v + 0.044715f * x3)));
    } else if (ACT == 2) {
      v = fmaxf(v, 0.f);
    }
    if (residual) v += residual[(size_t)(rbase + j) * N + col];
    C[(size_t)(rbase + j) * N + col] = v;
  }
}

// ---------------- attention: one block per (query s, head h) ----------------
__global__ __launch_bounds__(256)
void attn_kernel(const float* __restrict__ qkv, float* __restrict__ o) {
  const int s = blockIdx.x, h = blockIdx.y, tid = threadIdx.x;
  const int lane = tid & 31, wid = tid >> 5;
  __shared__ float q[kDH];
  __shared__ float prob[kS];
  __shared__ float red[8];

  if (tid < kDH) q[tid] = qkv[(size_t)s * 3 * kD + h * kDH + tid];
  __syncthreads();

  float score = -1.0e9f;
  if (tid <= s) {
    const float* krow = qkv + (size_t)tid * 3 * kD + kD + h * kDH;
    float d = 0.f;
#pragma unroll 8
    for (int e = 0; e < kDH; ++e) d += q[e] * krow[e];
    score = d * 0.125f;  // 1/sqrt(64)
  }

  float m = wred_max(score);
  if (lane == 0) red[wid] = m;
  __syncthreads();
  if (tid == 0) {
    float mm = red[0];
    for (int i = 1; i < 8; ++i) mm = fmaxf(mm, red[i]);
    red[0] = mm;
  }
  __syncthreads();
  const float gmax = red[0];
  const float e = __expf(score - gmax);
  __syncthreads();  // everyone has read gmax before red[] is reused

  float ssum = wred_sum(e);
  if (lane == 0) red[wid] = ssum;
  __syncthreads();
  if (tid == 0) {
    float t = 0.f;
    for (int i = 0; i < 8; ++i) t += red[i];
    red[0] = t;
  }
  __syncthreads();
  prob[tid] = e / red[0];
  __syncthreads();

  if (tid < kDH) {
    const float* vcol = qkv + 2 * kD + h * kDH + tid;
    float accv = 0.f;
    for (int j = 0; j <= s; ++j) accv += prob[j] * vcol[(size_t)j * 3 * kD];
    o[(size_t)s * kD + h * kDH + tid] = accv;
  }
}

// ---------------- repack lat (fp32) -> lat16 (f16, A-fragment lane order) ----
// lat16[rt][kstep][lane][16 halves]; one wave per (kstep, rt).
__global__ __launch_bounds__(32)
void pack_a_kernel(const float* __restrict__ lat, _Float16* __restrict__ lat16) {
  const int kstep = blockIdx.x, rt = blockIdx.y;
  const int lane  = threadIdx.x & 31;
  v16h f = load_a_frag(lat, kD, rt * 16, kstep * 32);
  *(v16h*)(lat16 + (((size_t)rt * kKS + kstep) * 32 + lane) * 16) = f;
}

// ---------------- cm kernel: cm[n][r][p] = sum_v (lat[n].W_col + b)^2 ----------------
// Streams l2tt_w exactly once (float2 loads, 8B-aligned despite the +2 column
// offset of the core block). Each block stages a 768x16 f16 slice in LDS
// (column-major -> contiguous ds_load_b128 B-frags), sweeps all 16 row-tiles
// of pre-packed f16 lat fragments with WMMA (no per-iteration converts),
// squares and ds-accumulates into an LDS cm array. Branchless tail handling.
static constexpr int kNT = (kRVR + 15) / 16;  // 12565 column tiles

__global__ void __launch_bounds__(128)
cm_wmma(const _Float16* __restrict__ lat16, const float* __restrict__ Wtt,
        const float* __restrict__ btt, float* __restrict__ cm) {
  __shared__ __align__(32) _Float16 sW[kD * 16];  // 24 KB, column-major [c][k]
  __shared__ float scm[kS * 4 + 1];               // + dummy slot for tail lanes
  const int tid  = threadIdx.x;
  const int wave = tid >> 5;
  const int lane = tid & 31;

  for (int i = tid; i < kS * 4 + 1; i += 128) scm[i] = 0.f;
  __syncthreads();

  for (int tile = blockIdx.x; tile < kNT; tile += gridDim.x) {
    const int colbase = tile * 16;
    // stage 768x16 fp32 slice -> f16 LDS via float2 (global_load_b64);
    // clamp addresses + select values for the out-of-range tail columns.
    for (int i = tid; i < kD * 8; i += 128) {
      const int k  = i >> 3;             // 0..767
      const int c2 = (i & 7) << 1;       // 0,2,..,14
      const int gc  = colbase + c2;
      const int gcc = (gc + 1 < kRVR) ? gc : (kRVR - 2);
      float2 w2 = *(const float2*)(&Wtt[(size_t)k * kTT + 2 + gcc]);
      if (gc >= kRVR)     w2.x = 0.f;
      if (gc + 1 >= kRVR) w2.y = 0.f;
      sW[c2 * kD + k]       = (_Float16)w2.x;
      sW[(c2 + 1) * kD + k] = (_Float16)w2.y;
    }
    __syncthreads();

    const int   gc     = colbase + (lane & 15);
    const bool  cvalid = (gc < kRVR);
    const float bv     = btt[2 + (cvalid ? gc : (kRVR - 1))];
    const int   q      = ((gc >= kVR) ? 2 : 0) + (gc & 1);  // r*2 + p

    for (int rt = wave; rt < 16; rt += 4) {
      v8f acc = {0.f, 0.f, 0.f, 0.f, 0.f, 0.f, 0.f, 0.f};
#pragma unroll 4
      for (int ks = 0; ks < kKS; ++ks) {
        v16h a = load_a_packed(lat16, rt, ks);
        v16h b = load_b_frag_lds(sW, ks * 32);
        acc = __builtin_amdgcn_wmma_f32_16x16x32_f16(false, a, false, b,
                                                     (short)0, acc, false, false);
      }
      const int rbase = rt * 16 + ((lane >> 4) << 3);
#pragma unroll
      for (int j = 0; j < 8; ++j) {
        const float t = acc[j] + bv;
        const int slot = cvalid ? ((rbase + j) * 4 + q) : (kS * 4);
        atomicAdd(&scm[slot], t * t);  // ds_add_f32
      }
    }
    __syncthreads();
  }

  for (int i = tid; i < kTP * 4; i += 128)
    atomicAdd(&cm[i], scm[i]);
}

// ---------------- alpha/beta + gathered core slices ----------------
__global__ __launch_bounds__(256)
void ct_ab_kernel(const float* __restrict__ lat, const float* __restrict__ Wtt,
                  const float* __restrict__ btt, const int* __restrict__ ids,
                  float* __restrict__ ct, float* __restrict__ ab) {
  const int n = blockIdx.x, tid = threadIdx.x;
  const int lane = tid & 31, wid = tid >> 5;
  __shared__ int   cols[36];
  __shared__ float red[36 * 8];

  if (tid < 36) {
    int c;
    if (tid < 2)      c = tid;                       // alpha cols
    else if (tid < 4) c = kR + kRVR + (tid - 2);     // beta cols
    else {
      const int i = tid - 4, t = i >> 2, rp = i & 3;
      const int r = rp >> 1, pp = rp & 1;
      const int tok = ids[n + 1 + t];
      c = kR + r * kVR + tok * kR + pp;              // core[r, tok, p]
    }
    cols[tid] = c;
  }
  __syncthreads();

  int rc[36];
#pragma unroll
  for (int i = 0; i < 36; ++i) rc[i] = cols[i];

  float part[36];
#pragma unroll
  for (int i = 0; i < 36; ++i) part[i] = 0.f;
  for (int k = tid; k < kD; k += 256) {
    const float a = lat[(size_t)n * kD + k];
    const float* wr = Wtt + (size_t)k * kTT;
#pragma unroll
    for (int i = 0; i < 36; ++i) part[i] += a * wr[rc[i]];
  }
#pragma unroll
  for (int i = 0; i < 36; ++i) {
    float v = wred_sum(part[i]);
    if (lane == 0) red[i * 8 + wid] = v;
  }
  __syncthreads();
  if (tid < 36) {
    float s = 0.f;
    for (int w = 0; w < 8; ++w) s += red[tid * 8 + w];
    s += btt[cols[tid]];
    s = s * s;
    if (tid < 4) ab[n * 4 + tid] = s;          // [a0 a1 b0 b1]
    else         ct[n * 32 + (tid - 4)] = s;   // [t][r][p]
  }
}

// ---------------- final TT recursion + mean ----------------
__global__ __launch_bounds__(256)
void final_kernel(const float* __restrict__ cm, const float* __restrict__ ct,
                  const float* __restrict__ ab, float* __restrict__ out) {
  const int tid = threadIdx.x;
  __shared__ float red[8];
  float val = 0.f;
  if (tid < kTP) {
    const int n = tid;
    const float* A = ab + n * 4;
    float v0 = A[0], v1 = A[1], ln_num = 0.f;
#pragma unroll
    for (int t = 0; t < kHOR; ++t) {
      const float* c = ct + n * 32 + t * 4;
      float w0 = v0 * c[0] + v1 * c[2];
      float w1 = v0 * c[1] + v1 * c[3];
      float sf = w0 + w1;
      v0 = w0 / sf; v1 = w1 / sf;
      ln_num += logf(sf + kEPS);
    }
    ln_num += logf(v0 * A[2] + v1 * A[3] + kEPS);

    const float* m = cm + n * 4;
    v0 = A[0]; v1 = A[1];
    float ln_den = 0.f;
#pragma unroll
    for (int t = 0; t < kHOR; ++t) {
      float w0 = v0 * m[0] + v1 * m[2];
      float w1 = v0 * m[1] + v1 * m[3];
      float sf = w0 + w1;
      v0 = w0 / sf; v1 = w1 / sf;
      ln_den += logf(sf + kEPS);
    }
    ln_den += logf(v0 * A[2] + v1 * A[3] + kEPS);
    val = ln_den - ln_num;
  }
  float s = wred_sum(val);
  if ((tid & 31) == 0) red[tid >> 5] = s;
  __syncthreads();
  if (tid == 0) {
    float t = 0.f;
    for (int i = 0; i < 8; ++i) t += red[i];
    out[0] = t / (float)kTP;
  }
}

// ---------------- launcher ----------------
extern "C" void kernel_launch(void* const* d_in, const int* in_sizes, int n_in,
                              void* d_out, int out_size, void* d_ws, size_t ws_size,
                              hipStream_t stream) {
  const int*   ids    = (const int*)  d_in[0];
  const float* wte    = (const float*)d_in[1];
  const float* wpe    = (const float*)d_in[2];
  const float* ln1_g  = (const float*)d_in[3];
  const float* ln1_b  = (const float*)d_in[4];
  const float* qkv_w  = (const float*)d_in[5];
  const float* qkv_b  = (const float*)d_in[6];
  const float* proj_w = (const float*)d_in[7];
  const float* proj_b = (const float*)d_in[8];
  const float* ln2_g  = (const float*)d_in[9];
  const float* ln2_b  = (const float*)d_in[10];
  const float* fc1_w  = (const float*)d_in[11];
  const float* fc1_b  = (const float*)d_in[12];
  const float* fc2_w  = (const float*)d_in[13];
  const float* fc2_b  = (const float*)d_in[14];
  const float* lnf_g  = (const float*)d_in[15];
  const float* lnf_b  = (const float*)d_in[16];
  const float* s2l_w  = (const float*)d_in[17];
  const float* s2l_b  = (const float*)d_in[18];
  const float* l2tt_w = (const float*)d_in[19];
  const float* l2tt_b = (const float*)d_in[20];

  float* ws  = (float*)d_ws;
  float* h   = ws;                                // 256*768
  float* x   = h   + (size_t)kS * kD;             // 256*768
  float* qkv = x   + (size_t)kS * kD;             // 256*2304
  float* o   = qkv + (size_t)kS * 3 * kD;         // 256*768
  float* f1  = o   + (size_t)kS * kD;             // 256*3072
  float* lat = f1  + (size_t)kS * 4 * kD;         // 256*768 (rows 248..255 pad)
  float* cm  = lat + (size_t)kS * kD;             // 248*4
  float* ct  = cm  + (size_t)kTP * 4;             // 248*32
  float* ab  = ct  + (size_t)kTP * 32;            // 248*4
  _Float16* lat16 = (_Float16*)(ab + (size_t)kTP * 4);  // 16*24*32*16 halves

  embed_kernel<<<kS, 256, 0, stream>>>(ids, wte, wpe, h);

  for (int l = 0; l < kL; ++l) {
    ln_kernel<<<kS, 256, 0, stream>>>(h, ln1_g + l * kD, ln1_b + l * kD, x);
    gemm_wmma<0><<<dim3(3 * kD / 64, kS / 16), 128, 0, stream>>>(
        x, qkv_w + (size_t)l * kD * 3 * kD, qkv_b + l * 3 * kD, nullptr,
        qkv, 3 * kD, kD);
    attn_kernel<<<dim3(kS, kNH), 256, 0, stream>>>(qkv, o);
    gemm_wmma<0><<<dim3(kD / 64, kS / 16), 128, 0, stream>>>(
        o, proj_w + (size_t)l * kD * kD, proj_b + l * kD, h, h, kD, kD);
    ln_kernel<<<kS, 256, 0, stream>>>(h, ln2_g + l * kD, ln2_b + l * kD, x);
    gemm_wmma<1><<<dim3(4 * kD / 64, kS / 16), 128, 0, stream>>>(
        x, fc1_w + (size_t)l * kD * 4 * kD, fc1_b + l * 4 * kD, nullptr,
        f1, 4 * kD, kD);
    gemm_wmma<0><<<dim3(kD / 64, kS / 16), 128, 0, stream>>>(
        f1, fc2_w + (size_t)l * 4 * kD * kD, fc2_b + l * kD, h, h, kD, 4 * kD);
  }

  ln_kernel<<<kS, 256, 0, stream>>>(h, lnf_g, lnf_b, x);
  // writes all 256 rows; rows 248..255 are deterministic padding inside lat
  gemm_wmma<2><<<dim3(kD / 64, kS / 16), 128, 0, stream>>>(
      x, s2l_w, s2l_b, nullptr, lat, kD, kD);
  pack_a_kernel<<<dim3(kKS, 16), 32, 0, stream>>>(lat, lat16);

  hipMemsetAsync(cm, 0, (size_t)kTP * 4 * sizeof(float), stream);
  cm_wmma<<<1024, 128, 0, stream>>>(lat16, l2tt_w, l2tt_b, cm);
  ct_ab_kernel<<<kTP, 256, 0, stream>>>(lat, l2tt_w, l2tt_b, ids, ct, ab);
  final_kernel<<<1, 256, 0, stream>>>(cm, ct, ab, (float*)d_out);
}
